// TransformerEncoderLayer_82892868813531
// MI455X (gfx1250) — compile-verified, matrix operand order checked by reading, not run on hardware
//
#include <hip/hip_runtime.h>

// ---------------------------------------------------------------------------
// Bidirectional Mamba encoder layer for MI455X (gfx1250, wave32).
//   N=512, B=8, D=256  ->  8 sequences of length L=131072, D_INNER=2, D_STATE=16
//   Scan parallelized as chunked associative scan (3 phases).
//   FFN (256->2048->256) via v_wmma_f32_16x16x32_f16, A-tiles staged in LDS
//   with global_load_async_to_lds_b128 (ASYNCcnt path).
// ---------------------------------------------------------------------------

#define L_SEQ   131072
#define NB      8
#define CHUNK   512
#define NCHUNK  256
#define NCH_TOT 8192      // 2 dirs * 8 batch * 2 chan * 256 chunks
#define NROWS   4096      // N*B
#define DMODEL  256
#define DFFN    2048
#define EPS     1e-5f

typedef __attribute__((ext_vector_type(16))) _Float16 v16h;
typedef __attribute__((ext_vector_type(8)))  _Float16 v8h;
typedef __attribute__((ext_vector_type(8)))  float    v8f;
typedef __attribute__((ext_vector_type(4)))  int      v4i;

__device__ __forceinline__ float siluf(float x)     { return x * (1.0f / (1.0f + __expf(-x))); }
__device__ __forceinline__ float softplusf(float x) { return (x > 20.0f) ? x : log1pf(__expf(x)); }

// ---- async global->LDS copy (CDNA5 path), with graceful fallback ----------
#if defined(__has_builtin)
#  if __has_builtin(__builtin_amdgcn_global_load_async_to_lds_b128)
#    define HAVE_ASYNC_LDS 1
#  endif
#endif
#ifndef HAVE_ASYNC_LDS
#  define HAVE_ASYNC_LDS 0
#endif

__device__ __forceinline__ void async_cp16(const _Float16* g, _Float16* l)
{
#if HAVE_ASYNC_LDS
  // signature (from clang diagnostic): (v4i addrspace(1)*, v4i addrspace(3)*, imm, imm)
  __builtin_amdgcn_global_load_async_to_lds_b128(
      (__attribute__((address_space(1))) v4i*)g,
      (__attribute__((address_space(3))) v4i*)l, 0, 0);
#else
  *(v8h*)l = *(const v8h*)g;    // global_load_b128 + ds_store_b128
#endif
}

__device__ __forceinline__ void async_join()
{
#if HAVE_ASYNC_LDS
#  if __has_builtin(__builtin_amdgcn_s_wait_asynccnt)
  __builtin_amdgcn_s_wait_asynccnt(0);
#  else
  asm volatile("s_wait_asynccnt 0" ::: "memory");
#  endif
#endif
}

// ---------------------------------------------------------------------------
// Kernel 1: in-proj + causal depthwise conv(4) + SiLU + dt (softplus)
// ---------------------------------------------------------------------------
__global__ void mamba_pre(const float* __restrict__ src,
    const float* __restrict__ inw_f, const float* __restrict__ cw_f, const float* __restrict__ cb_f,
    const float* __restrict__ xp_f,  const float* __restrict__ dw_f, const float* __restrict__ dbv_f,
    const float* __restrict__ inw_b, const float* __restrict__ cw_b, const float* __restrict__ cb_b,
    const float* __restrict__ xp_b,  const float* __restrict__ dw_b, const float* __restrict__ dbv_b,
    float* __restrict__ xsA, float* __restrict__ zsA, float* __restrict__ dtA)
{
  int i = blockIdx.x * blockDim.x + threadIdx.x;
  if (i >= 2 * NB * L_SEQ) return;
  int t   = i % L_SEQ;
  int b   = (i / L_SEQ) % NB;
  int dir = i / (NB * L_SEQ);

  const float* inw = dir ? inw_b : inw_f;
  const float* cw  = dir ? cw_b  : cw_f;
  const float* cb  = dir ? cb_b  : cb_f;
  const float* xp  = dir ? xp_b  : xp_f;
  const float* dw  = dir ? dw_b  : dw_f;
  const float* dbv = dir ? dbv_b : dbv_f;

  float xv[4];
#pragma unroll
  for (int k = 0; k < 4; ++k) {
    int tt = t - 3 + k;
    xv[k] = (tt < 0) ? 0.0f : src[(size_t)b * L_SEQ + (size_t)(dir ? (L_SEQ - 1 - tt) : tt)];
  }

  float xs[2], zs[2];
#pragma unroll
  for (int c = 0; c < 2; ++c) {
    float acc = cb[c];
#pragma unroll
    for (int k = 0; k < 4; ++k) acc += cw[c * 4 + k] * inw[c] * xv[k];
    xs[c] = siluf(acc);
    zs[c] = siluf(xv[3] * inw[2 + c]);
  }

  size_t base = (size_t)((dir * NB + b) * 2) * L_SEQ;
  float dbc0 = xs[0] * xp[0] + xs[1] * xp[1];
#pragma unroll
  for (int c = 0; c < 2; ++c) {
    xsA[base + (size_t)c * L_SEQ + t] = xs[c];
    zsA[base + (size_t)c * L_SEQ + t] = zs[c];
    dtA[base + (size_t)c * L_SEQ + t] = softplusf(dbc0 * dw[c] + dbv[c]);
  }
}

// ---------------------------------------------------------------------------
// Kernel 2 (scan phase 1): per-chunk aggregates; one LANE per chunk, 16 states
// in registers.  h_out = P*h_in + S.
// ---------------------------------------------------------------------------
__global__ void scan_phase1(const float* __restrict__ xsA, const float* __restrict__ dtA,
    const float* __restrict__ xp_f, const float* __restrict__ Al_f,
    const float* __restrict__ xp_b, const float* __restrict__ Al_b,
    float* __restrict__ cP, float* __restrict__ cS)
{
  int gc = blockIdx.x * blockDim.x + threadIdx.x;
  if (gc >= NCH_TOT) return;
  int chunk = gc % NCHUNK;
  int c     = (gc / NCHUNK) & 1;
  int b     = (gc / (NCHUNK * 2)) % NB;
  int dir   =  gc / (NCHUNK * 2 * NB);

  const float* xp = dir ? xp_b : xp_f;
  const float* Al = dir ? Al_b : Al_f;

  float Aa[16], xb0[16], xb1[16], P[16], S[16];
#pragma unroll
  for (int s = 0; s < 16; ++s) {
    Aa[s]  = -__expf(Al[c * 16 + s]);
    xb0[s] = xp[(1 + s) * 2 + 0];
    xb1[s] = xp[(1 + s) * 2 + 1];
    P[s] = 1.0f; S[s] = 0.0f;
  }

  size_t base0 = (size_t)((dir * NB + b) * 2) * L_SEQ;
  size_t basec = base0 + (size_t)c * L_SEQ;
  int t0 = chunk * CHUNK;
  for (int tt = 0; tt < CHUNK; ++tt) {
    int t = t0 + tt;
    float x0 = xsA[base0 + t];
    float x1 = xsA[base0 + L_SEQ + t];
    float d  = dtA[basec + t];
    float xc = c ? x1 : x0;
    float dx = d * xc;
#pragma unroll
    for (int s = 0; s < 16; ++s) {
      float a = __expf(d * Aa[s]);
      float u = dx * (x0 * xb0[s] + x1 * xb1[s]);
      S[s] = S[s] * a + u;
      P[s] = P[s] * a;
    }
  }
#pragma unroll
  for (int s = 0; s < 16; ++s) {
    cP[(size_t)gc * 16 + s] = P[s];
    cS[(size_t)gc * 16 + s] = S[s];
  }
}

// ---------------------------------------------------------------------------
// Kernel 3 (scan phase 2): scan the 256 chunk aggregates per (dir,b,c,s).
// ---------------------------------------------------------------------------
__global__ void scan_phase2(const float* __restrict__ cP, const float* __restrict__ cS,
                            float* __restrict__ carry)
{
  int id = blockIdx.x * blockDim.x + threadIdx.x;
  if (id >= 2 * NB * 2 * 16) return;          // 512 channels
  int s       = id & 15;
  int ch_base = id >> 4;                      // (dir,b,c) in 0..31
  float h = 0.0f;
  for (int ch = 0; ch < NCHUNK; ++ch) {
    size_t g = (size_t)(ch_base * NCHUNK + ch) * 16 + s;
    carry[g] = h;
    h = h * cP[g] + cS[g];
  }
}

// ---------------------------------------------------------------------------
// Kernel 4 (scan phase 3): re-walk each chunk from its carry, emit y_part.
// ---------------------------------------------------------------------------
__global__ void scan_phase3(const float* __restrict__ xsA, const float* __restrict__ zsA,
    const float* __restrict__ dtA, const float* __restrict__ carry,
    const float* __restrict__ xp_f, const float* __restrict__ Al_f,
    const float* __restrict__ Dp_f, const float* __restrict__ ow_f,
    const float* __restrict__ xp_b, const float* __restrict__ Al_b,
    const float* __restrict__ Dp_b, const float* __restrict__ ow_b,
    float* __restrict__ yp)
{
  int gc = blockIdx.x * blockDim.x + threadIdx.x;
  if (gc >= NCH_TOT) return;
  int chunk = gc % NCHUNK;
  int c     = (gc / NCHUNK) & 1;
  int b     = (gc / (NCHUNK * 2)) % NB;
  int dir   =  gc / (NCHUNK * 2 * NB);

  const float* xp = dir ? xp_b : xp_f;
  const float* Al = dir ? Al_b : Al_f;
  float dp = (dir ? Dp_b : Dp_f)[c];
  float ow = (dir ? ow_b : ow_f)[c];

  float Aa[16], xb0[16], xb1[16], xc0[16], xc1[16], h[16];
#pragma unroll
  for (int s = 0; s < 16; ++s) {
    Aa[s]  = -__expf(Al[c * 16 + s]);
    xb0[s] = xp[(1 + s) * 2 + 0];
    xb1[s] = xp[(1 + s) * 2 + 1];
    xc0[s] = xp[(17 + s) * 2 + 0];
    xc1[s] = xp[(17 + s) * 2 + 1];
    h[s]   = carry[(size_t)gc * 16 + s];
  }

  size_t base0 = (size_t)((dir * NB + b) * 2) * L_SEQ;
  size_t basec = base0 + (size_t)c * L_SEQ;
  size_t ybase = (size_t)(dir * 2 + c) * (NB * (size_t)L_SEQ) + (size_t)b * L_SEQ;
  int t0 = chunk * CHUNK;
  for (int tt = 0; tt < CHUNK; ++tt) {
    int t = t0 + tt;
    float x0 = xsA[base0 + t];
    float x1 = xsA[base0 + L_SEQ + t];
    float d  = dtA[basec + t];
    float xc = c ? x1 : x0;
    float dx = d * xc;
    float y = 0.0f;
#pragma unroll
    for (int s = 0; s < 16; ++s) {
      float a = __expf(d * Aa[s]);
      float u = dx * (x0 * xb0[s] + x1 * xb1[s]);
      h[s] = h[s] * a + u;
      y += h[s] * (x0 * xc0[s] + x1 * xc1[s]);
    }
    float zf = zsA[basec + t];
    float yf = (y + dp * xc) * zf * ow;
    int tpos = dir ? (L_SEQ - 1 - t) : t;
    yp[ybase + tpos] = yf;
  }
}

// ---------------------------------------------------------------------------
// Kernel 5: s = LayerNorm(src + sum of 4 y_part buffers); emits f32 and f16.
// ---------------------------------------------------------------------------
__global__ void ln1_kernel(const float* __restrict__ src, const float* __restrict__ yp,
                           const float* __restrict__ g, const float* __restrict__ be,
                           float* __restrict__ sF, _Float16* __restrict__ sH)
{
  __shared__ float sm[DMODEL];
  int row = blockIdx.x;
  int d   = threadIdx.x;
  size_t i = (size_t)row * DMODEL + d;
  size_t P = (size_t)NB * L_SEQ;
  float v = src[i] + yp[i] + yp[P + i] + yp[2 * P + i] + yp[3 * P + i];

  sm[d] = v; __syncthreads();
  for (int off = DMODEL / 2; off > 0; off >>= 1) { if (d < off) sm[d] += sm[d + off]; __syncthreads(); }
  float mean = sm[0] * (1.0f / DMODEL); __syncthreads();
  float dv = v - mean;
  sm[d] = dv * dv; __syncthreads();
  for (int off = DMODEL / 2; off > 0; off >>= 1) { if (d < off) sm[d] += sm[d + off]; __syncthreads(); }
  float var = sm[0] * (1.0f / DMODEL); __syncthreads();
  float o = dv * rsqrtf(var + EPS) * g[d] + be[d];
  sF[i] = o;
  sH[i] = (_Float16)o;
}

// ---------------------------------------------------------------------------
// Kernel 6: convert FFN weights f32 -> f16.
// ---------------------------------------------------------------------------
__global__ void cvt_w(const float* __restrict__ w1, const float* __restrict__ w2,
                      _Float16* __restrict__ w1h, _Float16* __restrict__ w2h)
{
  int i = blockIdx.x * blockDim.x + threadIdx.x;
  const int n = DFFN * DMODEL;               // 524288 each
  if (i < n)           w1h[i]     = (_Float16)w1[i];
  else if (i < 2 * n)  w2h[i - n] = (_Float16)w2[i - n];
}

// ---------------------------------------------------------------------------
// WMMA fragment loader (A/B 16x32 f16 fragment per ISA 7.12.2):
// lane<16: row m=lane, K {kb+0..7, kb+16..23}; lane>=16: K {kb+8..15, kb+24..31}
// Works for both global and LDS source pointers (generic addressing).
// ---------------------------------------------------------------------------
__device__ __forceinline__ v16h load_frag(const _Float16* rowptr, int kb, int grp)
{
  union { v16h v; v8h h[2]; } u;
  u.h[0] = *(const v8h*)(rowptr + kb + grp * 8);
  u.h[1] = *(const v8h*)(rowptr + kb + 16 + grp * 8);
  return u.v;
}

// ---------------------------------------------------------------------------
// Kernel 7: GEMM1  hidden = relu(s @ w1^T + b1)   (4096x256)x(256x2048)
// A tile (16 x 256 halves = 8KB) shared by all 8 waves of the block: staged
// into LDS with async copy; B streams from global (unique per wave).
// ---------------------------------------------------------------------------
__global__ void gemm1_wmma(const _Float16* __restrict__ Ah, const _Float16* __restrict__ Bh,
                           const float* __restrict__ bias, _Float16* __restrict__ H)
{
  __shared__ _Float16 smA[16 * DMODEL];     // 8KB
  int wave = threadIdx.x >> 5;
  int lane = threadIdx.x & 31;
  int tm   = blockIdx.x >> 4;               // 16 blocks per row of 128 N-tiles
  int tn   = (blockIdx.x & 15) * 8 + wave;

  // cooperative async A-tile copy: 512 x 16B chunks
  for (int ci = threadIdx.x; ci < (16 * DMODEL) / 8; ci += 256) {
    int row = ci >> 5;                      // 32 chunks per row
    int off = (ci & 31) * 8;                // halves
    async_cp16(Ah + (size_t)(tm * 16 + row) * DMODEL + off, &smA[row * DMODEL + off]);
  }
  async_join();
  __syncthreads();

  int grp = lane >> 4, m = lane & 15;
  const _Float16* arow = &smA[m * DMODEL];
  const _Float16* brow = Bh + (size_t)(tn * 16 + m) * DMODEL;
  v8f acc = {};
  for (int kb = 0; kb < DMODEL; kb += 32) {
    __builtin_prefetch(brow + kb + 32, 0, 1);
    v16h a = load_frag(arow, kb, grp);      // ds_load_b128 x2
    v16h b = load_frag(brow, kb, grp);      // global_load_b128 x2
    acc = __builtin_amdgcn_wmma_f32_16x16x32_f16(false, a, false, b, (short)0, acc, false, false);
  }
  int col   = tn * 16 + m;
  float bv  = bias[col];
  int rbase = tm * 16 + grp * 8;
#pragma unroll
  for (int r = 0; r < 8; ++r) {
    float v = acc[r] + bv;
    v = v > 0.0f ? v : 0.0f;
    H[(size_t)(rbase + r) * DFFN + col] = (_Float16)v;
  }
}

// ---------------------------------------------------------------------------
// Kernel 8: GEMM2  ffn_pre = s + (hidden @ w2^T + b2)   (4096x2048)x(2048x256)
// A tile = 16 x 2048 halves = 64KB in LDS (320KB WGP budget).
// ---------------------------------------------------------------------------
__global__ void gemm2_wmma(const _Float16* __restrict__ Ah, const _Float16* __restrict__ Bh,
                           const float* __restrict__ bias, const float* __restrict__ sF,
                           float* __restrict__ ffn_pre)
{
  __shared__ _Float16 smA[16 * DFFN];       // 64KB
  int wave = threadIdx.x >> 5;
  int lane = threadIdx.x & 31;
  int tm   = blockIdx.x >> 1;               // 2 blocks per row of 16 N-tiles
  int tn   = (blockIdx.x & 1) * 8 + wave;

  // cooperative async A-tile copy: 4096 x 16B chunks
  for (int ci = threadIdx.x; ci < (16 * DFFN) / 8; ci += 256) {
    int row = ci >> 8;                      // 256 chunks per row
    int off = (ci & 255) * 8;               // halves
    async_cp16(Ah + (size_t)(tm * 16 + row) * DFFN + off, &smA[row * DFFN + off]);
  }
  async_join();
  __syncthreads();

  int grp = lane >> 4, m = lane & 15;
  const _Float16* arow = &smA[m * DFFN];
  const _Float16* brow = Bh + (size_t)(tn * 16 + m) * DFFN;
  v8f acc = {};
  for (int kb = 0; kb < DFFN; kb += 32) {
    __builtin_prefetch(brow + kb + 32, 0, 1);
    v16h a = load_frag(arow, kb, grp);
    v16h b = load_frag(brow, kb, grp);
    acc = __builtin_amdgcn_wmma_f32_16x16x32_f16(false, a, false, b, (short)0, acc, false, false);
  }
  int col   = tn * 16 + m;
  float bv  = bias[col];
  int rbase = tm * 16 + grp * 8;
#pragma unroll
  for (int r = 0; r < 8; ++r) {
    size_t idx = (size_t)(rbase + r) * DMODEL + col;
    ffn_pre[idx] = sF[idx] + acc[r] + bv;
  }
}

// ---------------------------------------------------------------------------
// Kernel 9: final LayerNorm -> d_out (f32)
// ---------------------------------------------------------------------------
__global__ void ln2_kernel(const float* __restrict__ x, const float* __restrict__ g,
                           const float* __restrict__ be, float* __restrict__ out)
{
  __shared__ float sm[DMODEL];
  int row = blockIdx.x;
  int d   = threadIdx.x;
  size_t i = (size_t)row * DMODEL + d;
  float v = x[i];
  sm[d] = v; __syncthreads();
  for (int off = DMODEL / 2; off > 0; off >>= 1) { if (d < off) sm[d] += sm[d + off]; __syncthreads(); }
  float mean = sm[0] * (1.0f / DMODEL); __syncthreads();
  float dv = v - mean;
  sm[d] = dv * dv; __syncthreads();
  for (int off = DMODEL / 2; off > 0; off >>= 1) { if (d < off) sm[d] += sm[d + off]; __syncthreads(); }
  float var = sm[0] * (1.0f / DMODEL); __syncthreads();
  out[i] = dv * rsqrtf(var + EPS) * g[d] + be[d];
}

// ---------------------------------------------------------------------------
// Host launcher
// ---------------------------------------------------------------------------
extern "C" void kernel_launch(void* const* d_in, const int* in_sizes, int n_in,
                              void* d_out, int out_size, void* d_ws, size_t ws_size,
                              hipStream_t stream)
{
  const float* src   = (const float*)d_in[0];
  const float* inw_f = (const float*)d_in[1];
  const float* cw_f  = (const float*)d_in[2];
  const float* cb_f  = (const float*)d_in[3];
  const float* xp_f  = (const float*)d_in[4];
  const float* dw_f  = (const float*)d_in[5];
  const float* dbv_f = (const float*)d_in[6];
  const float* Al_f  = (const float*)d_in[7];
  const float* Dp_f  = (const float*)d_in[8];
  const float* ow_f  = (const float*)d_in[9];
  const float* inw_b = (const float*)d_in[10];
  const float* cw_b  = (const float*)d_in[11];
  const float* cb_b  = (const float*)d_in[12];
  const float* xp_b  = (const float*)d_in[13];
  const float* dw_b  = (const float*)d_in[14];
  const float* dbv_b = (const float*)d_in[15];
  const float* Al_b  = (const float*)d_in[16];
  const float* Dp_b  = (const float*)d_in[17];
  const float* ow_b  = (const float*)d_in[18];
  const float* w1    = (const float*)d_in[19];
  const float* b1    = (const float*)d_in[20];
  const float* w2    = (const float*)d_in[21];
  const float* b2    = (const float*)d_in[22];
  const float* g1    = (const float*)d_in[23];
  const float* be1   = (const float*)d_in[24];
  const float* g2    = (const float*)d_in[25];
  const float* be2   = (const float*)d_in[26];
  float* out = (float*)d_out;

  char* ws = (char*)d_ws;
  size_t off = 0;
  auto carve = [&](size_t bytes) -> char* {
    char* p = ws + off;
    off += (bytes + 255) & ~(size_t)255;
    return p;
  };
  const size_t CHAN = (size_t)2 * NB * 2 * L_SEQ;   // 4,194,304
  float*    xsA     = (float*)   carve(CHAN * 4);
  float*    zsA     = (float*)   carve(CHAN * 4);
  float*    dtA     = (float*)   carve(CHAN * 4);
  float*    cP      = (float*)   carve((size_t)NCH_TOT * 16 * 4);
  float*    cS      = (float*)   carve((size_t)NCH_TOT * 16 * 4);
  float*    carry   = (float*)   carve((size_t)NCH_TOT * 16 * 4);
  float*    yp      = (float*)   carve((size_t)4 * NB * L_SEQ * 4);
  float*    sF      = (float*)   carve((size_t)NROWS * DMODEL * 4);
  float*    ffn_pre = (float*)   carve((size_t)NROWS * DMODEL * 4);
  _Float16* sH      = (_Float16*)carve((size_t)NROWS * DMODEL * 2);
  _Float16* w1h     = (_Float16*)carve((size_t)DFFN * DMODEL * 2);
  _Float16* w2h     = (_Float16*)carve((size_t)DMODEL * DFFN * 2);
  _Float16* hid     = (_Float16*)carve((size_t)NROWS * DFFN * 2);
  (void)ws_size; (void)in_sizes; (void)n_in; (void)out_size;

  // 1) channel pipeline (both directions)
  mamba_pre<<<(2 * NB * L_SEQ) / 256, 256, 0, stream>>>(
      src, inw_f, cw_f, cb_f, xp_f, dw_f, dbv_f,
           inw_b, cw_b, cb_b, xp_b, dw_b, dbv_b, xsA, zsA, dtA);

  // 2) chunked associative scan
  scan_phase1<<<NCH_TOT / 256, 256, 0, stream>>>(xsA, dtA, xp_f, Al_f, xp_b, Al_b, cP, cS);
  scan_phase2<<<2, 256, 0, stream>>>(cP, cS, carry);
  scan_phase3<<<NCH_TOT / 256, 256, 0, stream>>>(xsA, zsA, dtA, carry,
      xp_f, Al_f, Dp_f, ow_f, xp_b, Al_b, Dp_b, ow_b, yp);

  // 3) residual + LN1 (also produces f16 activations for WMMA)
  ln1_kernel<<<NROWS, DMODEL, 0, stream>>>(src, yp, g1, be1, sF, sH);

  // 4) FFN via WMMA (A tiles staged through LDS with async copies)
  cvt_w<<<(2 * DFFN * DMODEL) / 256, 256, 0, stream>>>(w1, w2, w1h, w2h);
  gemm1_wmma<<<((NROWS / 16) * (DFFN / 16)) / 8, 256, 0, stream>>>(sH, w1h, b1, hid);
  gemm2_wmma<<<((NROWS / 16) * (DMODEL / 16)) / 8, 256, 0, stream>>>(hid, w2h, b2, sF, ffn_pre);

  // 5) residual + LN2 -> output
  ln2_kernel<<<NROWS, DMODEL, 0, stream>>>(ffn_pre, g2, be2, out);
}
